// GeoNetGraph_5007931867356
// MI455X (gfx1250) — matrix operand order repaired and checked by reading, not compiled, not run on hardware
//
#include <hip/hip_runtime.h>
#include <hip/hip_bf16.h>

#define NN 4096
#define HID 128
#define NHEADS 4
#define DH 32
#define NE 131072
#define INDIM 256
#define NOUT 40

typedef _Float16 half16 __attribute__((ext_vector_type(16)));
typedef _Float16 v8h    __attribute__((ext_vector_type(8)));
typedef float    f32x8  __attribute__((ext_vector_type(8)));

union H16U { half16 h; unsigned int u[8]; };

static __device__ inline f32x8 wmma_f16(half16 a, half16 b, f32x8 c) {
  return __builtin_amdgcn_wmma_f32_16x16x32_f16(false, a, false, b, (short)0, c, false, false);
}

// 16 contiguous halves (B-matrix per-lane layout): two 16B loads
static __device__ inline half16 ld16(const _Float16* p) {
  v8h lo = *(const v8h*)p;
  v8h hi = *(const v8h*)(p + 8);
  half16 r;
#pragma unroll
  for (int i = 0; i < 8; ++i) { r[i] = lo[i]; r[i + 8] = hi[i]; }
  return r;
}

// A-matrix per-lane layout: e0..15 = base[kh*16 .. +15]
static __device__ inline half16 ld_a(const _Float16* base, int kh) {
  v8h lo = *(const v8h*)(base + kh * 16);
  v8h hi = *(const v8h*)(base + kh * 16 + 8);
  half16 r;
#pragma unroll
  for (int i = 0; i < 8; ++i) { r[i] = lo[i]; r[i + 8] = hi[i]; }
  return r;
}

static __device__ inline unsigned int pkf16(float a, float b) {
  union { unsigned int u; _Float16 h[2]; } x;
  x.h[0] = (_Float16)a; x.h[1] = (_Float16)b;
  return x.u;
}

static __device__ inline float wred_sum(float v) {
#pragma unroll
  for (int m = 16; m >= 1; m >>= 1) v += __shfl_xor(v, m, 32);
  return v;
}

static __device__ inline float exp0_scale(float sumsq, float sc) {
  float n = fmaxf(sqrtf(sumsq), 1e-7f);
  return tanhf(sc * n) / (sc * n);
}
static __device__ inline float log0_scale(float sumsq, float sc) {
  float n = fmaxf(sqrtf(sumsq), 1e-7f);
  float t = fminf(sc * n, 1.f - 1e-5f);
  return 0.5f * __logf((1.f + t) / (1.f - t)) / (sc * n);
}

// ---------------- utility kernels ----------------

__global__ void zero_kernel(float* p, int n) {
  int i = blockIdx.x * blockDim.x + threadIdx.x;
  if (i < n) p[i] = 0.f;
}

__global__ void cvt_f16_kernel(const float* __restrict__ in, _Float16* __restrict__ out, int n) {
  int i = blockIdx.x * blockDim.x + threadIdx.x;
  if (i < n) out[i] = (_Float16)in[i];
}

// W is (K x ncols) row-major f32 -> Wt is (ncolsPad x K) f16, column-of-W contiguous, zero-padded
__global__ void wt_prep_kernel(const float* __restrict__ W, _Float16* __restrict__ Wt,
                               int K, int ncols, int ncolsPad) {
  int t = blockIdx.x * blockDim.x + threadIdx.x;
  if (t >= K * ncolsPad) return;
  int col = t / K, kk = t - col * K;
  float v = (col < ncols) ? W[(size_t)kk * ncols + col] : 0.f;
  Wt[(size_t)col * K + kk] = (_Float16)v;
}

__global__ void deg_kernel(const int* __restrict__ dst, float* __restrict__ deg, int E) {
  int e = blockIdx.x * blockDim.x + threadIdx.x;
  if (e < E) atomicAdd(&deg[dst[e]], 1.f);
}

__global__ void scatter_kernel(const int* __restrict__ src, const int* __restrict__ dst,
                               const float* __restrict__ m, float* __restrict__ agg, int E) {
  int t = blockIdx.x * blockDim.x + threadIdx.x;
  int e = t >> 5;
  if (e >= E) return;
  int d0 = (t & 31) * 4;
  const float* ms = m + (size_t)src[e] * HID + d0;
  float* ad = agg + (size_t)dst[e] * HID + d0;
#pragma unroll
  for (int i = 0; i < 4; ++i) atomicAdd(ad + i, ms[i]);
}

// ---------------- WMMA GEMM:  C(M x ncols) = A(M x K) * W(K x ncols) + bias ----------------
// One wave per 16-row block, NT column tiles. A f16 row-major, Wt f16 "col-major" transposed.
template <int NT>
__global__ void __launch_bounds__(256) gemm_wmma_kernel(
    const _Float16* __restrict__ A, const _Float16* __restrict__ Wt,
    const float* __restrict__ bias,
    float* __restrict__ outF32, int ldF32,
    _Float16* __restrict__ outF16, int ldF16,
    _Float16* __restrict__ outF16T, int ldT,
    int M, int K, int ncols, float scale, int relu) {
  const int lane = threadIdx.x & 31;
  const int rb = blockIdx.x * (blockDim.x >> 5) + (threadIdx.x >> 5);
  if (rb * 16 >= M) return;
  const int n0 = lane & 15;
  const int kh = lane >> 4;

  f32x8 acc[NT];
#pragma unroll
  for (int t = 0; t < NT; ++t) acc[t] = 0.f;

  const _Float16* arow = A + (size_t)(rb * 16 + n0) * K;
  for (int kc = 0; kc < K; kc += 32) {
    half16 a = ld_a(arow + kc, kh);
#pragma unroll
    for (int t = 0; t < NT; ++t) {
      half16 b = ld16(Wt + (size_t)(t * 16 + n0) * K + kc + kh * 16);
      acc[t] = wmma_f16(a, b, acc[t]);
    }
  }
#pragma unroll
  for (int t = 0; t < NT; ++t) {
    int col = t * 16 + n0;
    bool ok = col < ncols;
    float bv = ok ? bias[col] : 0.f;
#pragma unroll
    for (int r = 0; r < 8; ++r) {
      int row = rb * 16 + r + 8 * kh;
      float v = (acc[t][r] + bv) * scale;
      if (relu) v = fmaxf(v, 0.f);
      if (ok) {
        if (outF32)  outF32[(size_t)row * ldF32 + col] = v;
        if (outF16)  outF16[(size_t)row * ldF16 + col] = (_Float16)v;
        if (outF16T) outF16T[(size_t)col * ldT + row]  = (_Float16)v;
      }
    }
  }
}

// ---------------- flash attention (one wave per (16-query block, head)) ----------------
// Computes S^T = K_tile * Q^T so online-softmax stats are per-lane scalars (col = query row).
__global__ void __launch_bounds__(256) flash_attn_kernel(
    const _Float16* __restrict__ q, const _Float16* __restrict__ k,
    const _Float16* __restrict__ vt, _Float16* __restrict__ o) {
  const int lane = threadIdx.x & 31;
  const int wid = blockIdx.x * (blockDim.x >> 5) + (threadIdx.x >> 5);
  const int hd = wid & (NHEADS - 1);
  const int qb = wid >> 2;
  const int n0 = lane & 15;
  const int kh = lane >> 4;

  // Q^T as B-matrix: lane holds query row n0, dims kh*16..kh*16+15 (q already scaled by 1/sqrt(DH))
  const half16 Bq = ld16(q + (size_t)(qb * 16 + n0) * HID + hd * DH + kh * 16);

  f32x8 acc0 = 0.f, acc1 = 0.f;   // O^T dims 0..15 / 16..31 of this head
  float mrow = -3.0e38f, lrow = 0.f;

  for (int j = 0; j < NN; j += 32) {
    // K rows as A-matrix (M=key, K=dim)
    half16 A0 = ld_a(k + (size_t)(j + n0) * HID + hd * DH, kh);
    half16 A1 = ld_a(k + (size_t)(j + 16 + n0) * HID + hd * DH, kh);
    f32x8 s0 = 0.f, s1 = 0.f;
    s0 = wmma_f16(A0, Bq, s0);   // S^T keys j..j+15   (rows), qrows (cols)
    s1 = wmma_f16(A1, Bq, s1);   // S^T keys j+16..j+31

    float pm = -3.0e38f;
#pragma unroll
    for (int r = 0; r < 8; ++r) pm = fmaxf(pm, fmaxf(s0[r], s1[r]));
    pm = fmaxf(pm, __shfl_xor(pm, 16, 32));     // combine both key halves (same qrow)
    const float mnew = fmaxf(mrow, pm);
    const float corr = __expf(mrow - mnew);

    float p0[8], p1[8];
    float ps = 0.f;
#pragma unroll
    for (int r = 0; r < 8; ++r) {
      p0[r] = __expf(s0[r] - mnew);
      p1[r] = __expf(s1[r] - mnew);
      ps += p0[r] + p1[r];
    }
    ps += __shfl_xor(ps, 16, 32);
    lrow = lrow * corr + ps;
    mrow = mnew;
#pragma unroll
    for (int r = 0; r < 8; ++r) { acc0[r] *= corr; acc1[r] *= corr; }

    // repack P^T (C-layout) into B-layout with xor-16 shuffles
    unsigned int u0[4], u1[4];
#pragma unroll
    for (int t = 0; t < 4; ++t) {
      u0[t] = pkf16(p0[2 * t], p0[2 * t + 1]);
      u1[t] = pkf16(p1[2 * t], p1[2 * t + 1]);
    }
    H16U Bp;
#pragma unroll
    for (int t = 0; t < 4; ++t) {
      unsigned int o0 = (unsigned int)__shfl_xor((int)u0[t], 16, 32);
      unsigned int o1 = (unsigned int)__shfl_xor((int)u1[t], 16, 32);
      Bp.u[t]     = kh ? o1 : u0[t];
      Bp.u[t + 4] = kh ? u1[t] : o0;
    }

    // V^T as A-matrix (M=dim, K=key); vt is (HID x NN) f16, pre-transposed
    half16 Av0 = ld_a(vt + (size_t)(hd * DH + n0) * NN + j, kh);
    half16 Av1 = ld_a(vt + (size_t)(hd * DH + 16 + n0) * NN + j, kh);
    acc0 = wmma_f16(Av0, Bp.h, acc0);
    acc1 = wmma_f16(Av1, Bp.h, acc1);
  }
  const float inv = 1.f / lrow;
  _Float16* orow = o + (size_t)(qb * 16 + n0) * HID + hd * DH;
#pragma unroll
  for (int r = 0; r < 8; ++r) {
    orow[r + 8 * kh]      = (_Float16)(acc0[r] * inv);
    orow[16 + r + 8 * kh] = (_Float16)(acc1[r] * inv);
  }
}

// ---------------- per-row (wave-per-node) hyperbolic kernels ----------------

__global__ void __launch_bounds__(256) exp0_kernel(const float* __restrict__ in,
                                                   float* __restrict__ out,
                                                   const float* __restrict__ cptr) {
  const int lane = threadIdx.x & 31;
  const int row = blockIdx.x * (blockDim.x >> 5) + (threadIdx.x >> 5);
  const float sc = sqrtf(fabsf(*cptr));
  const float* r = in + (size_t)row * HID;
  float x[4]; float ss = 0.f;
#pragma unroll
  for (int i = 0; i < 4; ++i) { x[i] = r[lane + 32 * i]; ss += x[i] * x[i]; }
  float s = exp0_scale(wred_sum(ss), sc);
  float* w = out + (size_t)row * HID;
#pragma unroll
  for (int i = 0; i < 4; ++i) w[lane + 32 * i] = x[i] * s;
}

__global__ void __launch_bounds__(256) log0_kernel(const float* __restrict__ in,
                                                   float* __restrict__ out,
                                                   _Float16* __restrict__ out16,
                                                   const float* __restrict__ cptr) {
  const int lane = threadIdx.x & 31;
  const int row = blockIdx.x * (blockDim.x >> 5) + (threadIdx.x >> 5);
  const float sc = sqrtf(fabsf(*cptr));
  const float* r = in + (size_t)row * HID;
  float x[4]; float ss = 0.f;
#pragma unroll
  for (int i = 0; i < 4; ++i) { x[i] = r[lane + 32 * i]; ss += x[i] * x[i]; }
  float s = log0_scale(wred_sum(ss), sc);
#pragma unroll
  for (int i = 0; i < 4; ++i) {
    float v = x[i] * s;
    out[(size_t)row * HID + lane + 32 * i] = v;
    if (out16) out16[(size_t)row * HID + lane + 32 * i] = (_Float16)v;
  }
}

__global__ void __launch_bounds__(256) conv_post_kernel(const float* __restrict__ agg,
                                                        const float* __restrict__ m,
                                                        const float* __restrict__ deg,
                                                        float* __restrict__ hconv,
                                                        const float* __restrict__ cptr) {
  const int lane = threadIdx.x & 31;
  const int row = blockIdx.x * (blockDim.x >> 5) + (threadIdx.x >> 5);
  const float sc = sqrtf(fabsf(*cptr));
  const float dinv = 1.f / (deg[row] + 1.f);
  float x[4]; float ss = 0.f;
#pragma unroll
  for (int i = 0; i < 4; ++i) {
    int d = lane + 32 * i;
    x[i] = (agg[(size_t)row * HID + d] + m[(size_t)row * HID + d]) * dinv;
    ss += x[i] * x[i];
  }
  float s = exp0_scale(wred_sum(ss), sc);
#pragma unroll
  for (int i = 0; i < 4; ++i) hconv[(size_t)row * HID + lane + 32 * i] = x[i] * s;
}

// fuse three branches, layer-norm in tangent space, tangent ReLU -> ball
__global__ void __launch_bounds__(256) fuse_kernel(const float* __restrict__ ht,
                                                   const float* __restrict__ hc,
                                                   const float* __restrict__ ha,
                                                   const float* __restrict__ gamma,
                                                   const float* __restrict__ beta,
                                                   const float* __restrict__ cptr,
                                                   float* __restrict__ hout) {
  const int lane = threadIdx.x & 31;
  const int row = blockIdx.x * (blockDim.x >> 5) + (threadIdx.x >> 5);
  const float sc = sqrtf(fabsf(*cptr));
  float xt[4], xc[4], xa[4];
  float ssc = 0.f, ssa = 0.f;
#pragma unroll
  for (int i = 0; i < 4; ++i) {
    int d = lane + 32 * i;
    xt[i] = ht[(size_t)row * HID + d];
    xc[i] = hc[(size_t)row * HID + d];
    xa[i] = ha[(size_t)row * HID + d];
    ssc += xc[i] * xc[i];
    ssa += xa[i] * xa[i];
  }
  float s_c = log0_scale(wred_sum(ssc), sc);
  float s_a = log0_scale(wred_sum(ssa), sc);
  float u1[4]; float ss1 = 0.f;
#pragma unroll
  for (int i = 0; i < 4; ++i) { u1[i] = xt[i] + xc[i] * s_c + xa[i] * s_a; ss1 += u1[i] * u1[i]; }
  float se = exp0_scale(wred_sum(ss1), sc);
  float h2[4]; float ss2 = 0.f;
#pragma unroll
  for (int i = 0; i < 4; ++i) { h2[i] = u1[i] * se; ss2 += h2[i] * h2[i]; }
  float sl = log0_scale(wred_sum(ss2), sc);
  float u2[4]; float sum = 0.f, sq = 0.f;
#pragma unroll
  for (int i = 0; i < 4; ++i) { u2[i] = h2[i] * sl; sum += u2[i]; sq += u2[i] * u2[i]; }
  sum = wred_sum(sum); sq = wred_sum(sq);
  float mu = sum * (1.f / HID);
  float var = sq * (1.f / HID) - mu * mu;
  float rs = rsqrtf(var + 1e-5f);
  float un[4]; float ss3 = 0.f;
#pragma unroll
  for (int i = 0; i < 4; ++i) {
    int d = lane + 32 * i;
    un[i] = (u2[i] - mu) * rs * gamma[d] + beta[d];
    ss3 += un[i] * un[i];
  }
  float se2 = exp0_scale(wred_sum(ss3), sc);
  float h3[4]; float ss4 = 0.f;
#pragma unroll
  for (int i = 0; i < 4; ++i) { h3[i] = un[i] * se2; ss4 += h3[i] * h3[i]; }
  float sl2 = log0_scale(wred_sum(ss4), sc);
  float u3[4]; float ss5 = 0.f;
#pragma unroll
  for (int i = 0; i < 4; ++i) { u3[i] = fmaxf(h3[i] * sl2, 0.f); ss5 += u3[i] * u3[i]; }
  float se3 = exp0_scale(wred_sum(ss5), sc);
#pragma unroll
  for (int i = 0; i < 4; ++i) hout[(size_t)row * HID + lane + 32 * i] = u3[i] * se3;
}

// ---------------- host orchestration ----------------

static inline void* carve(char** p, size_t bytes) {
  void* r = *p;
  *p += (bytes + 255) & ~(size_t)255;
  return r;
}

extern "C" void kernel_launch(void* const* d_in, const int* in_sizes, int n_in,
                              void* d_out, int out_size, void* d_ws, size_t ws_size,
                              hipStream_t stream) {
  (void)in_sizes; (void)n_in; (void)out_size; (void)ws_size;
  const float* x     = (const float*)d_in[0];
  const int*   ei    = (const int*)d_in[1];
  const float* cptr  = (const float*)d_in[2];
  const float* encW  = (const float*)d_in[3];
  const float* encb  = (const float*)d_in[4];
  const float* convW = (const float*)d_in[5];
  const float* convb = (const float*)d_in[6];
  const float* Wq    = (const float*)d_in[7];
  const float* bq    = (const float*)d_in[8];
  const float* Wk    = (const float*)d_in[9];
  const float* bk    = (const float*)d_in[10];
  const float* Wv    = (const float*)d_in[11];
  const float* bv    = (const float*)d_in[12];
  const float* Wo    = (const float*)d_in[13];
  const float* bo    = (const float*)d_in[14];
  const float* gamma = (const float*)d_in[15];
  const float* beta  = (const float*)d_in[16];
  const float* hW1   = (const float*)d_in[17];
  const float* hb1   = (const float*)d_in[18];
  const float* hW2   = (const float*)d_in[19];
  const float* hb2   = (const float*)d_in[20];
  float* out = (float*)d_out;

  char* p = (char*)d_ws;
  _Float16* x16    = (_Float16*)carve(&p, (size_t)NN * INDIM * 2);
  _Float16* W16    = (_Float16*)carve(&p, (size_t)327680 * 2);
  float*    h      = (float*)carve(&p, (size_t)NN * HID * 4);
  float*    htan   = (float*)carve(&p, (size_t)NN * HID * 4);
  _Float16* htan16 = (_Float16*)carve(&p, (size_t)NN * HID * 2);
  float*    mbuf   = (float*)carve(&p, (size_t)NN * HID * 4);
  float*    agg    = (float*)carve(&p, (size_t)NN * HID * 4);
  float*    hconv  = (float*)carve(&p, (size_t)NN * HID * 4);
  float*    hattn  = (float*)carve(&p, (size_t)NN * HID * 4);
  float*    tmp    = (float*)carve(&p, (size_t)NN * HID * 4);
  _Float16* q16    = (_Float16*)carve(&p, (size_t)NN * HID * 2);
  _Float16* k16    = (_Float16*)carve(&p, (size_t)NN * HID * 2);
  _Float16* vt16   = (_Float16*)carve(&p, (size_t)NN * HID * 2);
  _Float16* o16    = (_Float16*)carve(&p, (size_t)NN * HID * 2);
  _Float16* t116   = (_Float16*)carve(&p, (size_t)NN * HID * 2);
  float*    deg    = (float*)carve(&p, (size_t)NN * 4);

  // weight sub-buffers (halves)
  _Float16* encWt  = W16;                 // 128 x 256
  _Float16* convWt = W16 + 32768;         // 3 x (128 x 128)
  _Float16* WqT    = W16 + 81920;
  _Float16* WkT    = W16 + 131072;
  _Float16* WvT    = W16 + 180224;
  _Float16* WoT    = W16 + 229376;
  _Float16* hW1T   = W16 + 278528;        // 128 x 128
  _Float16* hW2T   = W16 + 294912;        // 48(pad) x 128

  const int* src = ei;
  const int* dst = ei + NE;

  const int TB = 256;
  const int gW128 = (128 * 128 + TB - 1) / TB;

  // ---- weight prep ----
  wt_prep_kernel<<<(INDIM * HID + TB - 1) / TB, TB, 0, stream>>>(encW, encWt, INDIM, HID, HID);
  for (int i = 0; i < 3; ++i) {
    wt_prep_kernel<<<gW128, TB, 0, stream>>>(convW + (size_t)i * HID * HID, convWt + (size_t)i * HID * HID, HID, HID, HID);
    wt_prep_kernel<<<gW128, TB, 0, stream>>>(Wq + (size_t)i * HID * HID, WqT + (size_t)i * HID * HID, HID, HID, HID);
    wt_prep_kernel<<<gW128, TB, 0, stream>>>(Wk + (size_t)i * HID * HID, WkT + (size_t)i * HID * HID, HID, HID, HID);
    wt_prep_kernel<<<gW128, TB, 0, stream>>>(Wv + (size_t)i * HID * HID, WvT + (size_t)i * HID * HID, HID, HID, HID);
    wt_prep_kernel<<<gW128, TB, 0, stream>>>(Wo + (size_t)i * HID * HID, WoT + (size_t)i * HID * HID, HID, HID, HID);
  }
  wt_prep_kernel<<<gW128, TB, 0, stream>>>(hW1, hW1T, HID, HID, HID);
  wt_prep_kernel<<<(HID * 48 + TB - 1) / TB, TB, 0, stream>>>(hW2, hW2T, HID, NOUT, 48);

  cvt_f16_kernel<<<(NN * INDIM + TB - 1) / TB, TB, 0, stream>>>(x, x16, NN * INDIM);

  const int gGemm = NN / 16 / 8;   // 32 blocks, 8 waves each
  const int gRow  = NN / 8;        // 512 blocks, wave-per-row
  const float qscale = 0.17677669529663687f;  // 1/sqrt(DH)

  // ---- encoder ----
  gemm_wmma_kernel<8><<<gGemm, TB, 0, stream>>>(x16, encWt, encb, tmp, HID,
                                                (_Float16*)nullptr, 0, (_Float16*)nullptr, 0,
                                                NN, INDIM, HID, 1.f, 0);
  exp0_kernel<<<gRow, TB, 0, stream>>>(tmp, h, cptr);

  // ---- degrees (constant across layers) ----
  zero_kernel<<<(NN + TB - 1) / TB, TB, 0, stream>>>(deg, NN);
  deg_kernel<<<(NE + TB - 1) / TB, TB, 0, stream>>>(dst, deg, NE);

  for (int i = 0; i < 3; ++i) {
    log0_kernel<<<gRow, TB, 0, stream>>>(h, htan, htan16, cptr);

    // graph conv branch
    gemm_wmma_kernel<8><<<gGemm, TB, 0, stream>>>(htan16, convWt + (size_t)i * HID * HID,
                                                  convb + (size_t)i * HID, mbuf, HID,
                                                  (_Float16*)nullptr, 0, (_Float16*)nullptr, 0,
                                                  NN, HID, HID, 1.f, 0);
    zero_kernel<<<(NN * HID + TB - 1) / TB, TB, 0, stream>>>(agg, NN * HID);
    scatter_kernel<<<(NE * 32 + TB - 1) / TB, TB, 0, stream>>>(src, dst, mbuf, agg, NE);
    conv_post_kernel<<<gRow, TB, 0, stream>>>(agg, mbuf, deg, hconv, cptr);

    // attention branch: q (pre-scaled), k, v (transposed) projections
    gemm_wmma_kernel<8><<<gGemm, TB, 0, stream>>>(htan16, WqT + (size_t)i * HID * HID,
                                                  bq + (size_t)i * HID, (float*)nullptr, 0,
                                                  q16, HID, (_Float16*)nullptr, 0,
                                                  NN, HID, HID, qscale, 0);
    gemm_wmma_kernel<8><<<gGemm, TB, 0, stream>>>(htan16, WkT + (size_t)i * HID * HID,
                                                  bk + (size_t)i * HID, (float*)nullptr, 0,
                                                  k16, HID, (_Float16*)nullptr, 0,
                                                  NN, HID, HID, 1.f, 0);
    gemm_wmma_kernel<8><<<gGemm, TB, 0, stream>>>(htan16, WvT + (size_t)i * HID * HID,
                                                  bv + (size_t)i * HID, (float*)nullptr, 0,
                                                  (_Float16*)nullptr, 0, vt16, NN,
                                                  NN, HID, HID, 1.f, 0);
    flash_attn_kernel<<<(NN / 16) * NHEADS / 8, TB, 0, stream>>>(q16, k16, vt16, o16);
    gemm_wmma_kernel<8><<<gGemm, TB, 0, stream>>>(o16, WoT + (size_t)i * HID * HID,
                                                  bo + (size_t)i * HID, tmp, HID,
                                                  (_Float16*)nullptr, 0, (_Float16*)nullptr, 0,
                                                  NN, HID, HID, 1.f, 0);
    exp0_kernel<<<gRow, TB, 0, stream>>>(tmp, hattn, cptr);

    // fuse + LN + tangent ReLU
    fuse_kernel<<<gRow, TB, 0, stream>>>(htan, hconv, hattn,
                                         gamma + (size_t)i * HID, beta + (size_t)i * HID,
                                         cptr, h);
  }

  // ---- head ----
  log0_kernel<<<gRow, TB, 0, stream>>>(h, htan, htan16, cptr);
  gemm_wmma_kernel<8><<<gGemm, TB, 0, stream>>>(htan16, hW1T, hb1, (float*)nullptr, 0,
                                                t116, HID, (_Float16*)nullptr, 0,
                                                NN, HID, HID, 1.f, 1 /*relu*/);
  gemm_wmma_kernel<3><<<gGemm, TB, 0, stream>>>(t116, hW2T, hb2, out, NOUT,
                                                (_Float16*)nullptr, 0, (_Float16*)nullptr, 0,
                                                NN, HID, NOUT, 1.f, 0);
}
